// PNALSPELayer_29368986370543
// MI455X (gfx1250) — matrix operand order repaired and checked by reading, not compiled, not run on hardware
//
#include <hip/hip_runtime.h>
#include <hip/hip_bf16.h>
#include <math.h>

typedef __attribute__((ext_vector_type(16))) _Float16 v16h;
typedef __attribute__((ext_vector_type(8)))  float    v8f;

#define NN 20000
#define DD 64
#define EE 320000

// ---------------- WMMA fragment helpers (CDNA5 16x16x32 f16, wave32) ----------

// A-matrix 16x32 f16 layout (ISA 7.12.2): lane m=lane&15; lanes>=16 use K+8;
// vgpr g<4 -> K=2g..2g+1, g>=4 -> K=8+2g..  => two contiguous 8-half runs.
static __device__ __forceinline__ v16h wmma_ld_a_lds(const _Float16* base, int ldk,
                                                     int k0, int lane) {
  const int m = lane & 15, hi = lane >> 4;
  const _Float16* r = base + m * ldk + k0 + hi * 8;
  v16h a;
#pragma unroll
  for (int g = 0; g < 8; ++g) {
    const int K = (g < 4) ? (2 * g) : (8 + 2 * g);
    a[2 * g]     = r[K];
    a[2 * g + 1] = r[K + 1];
  }
  return a;
}

static __device__ __forceinline__ v16h wmma_ld_a_g(const _Float16* A, int row0, int ldk,
                                                   int k0, int nrows, int lane) {
  const int m = lane & 15, hi = lane >> 4;
  int row = row0 + m;
  if (row >= nrows) row = 0;  // dummy row; result discarded
  const _Float16* r = A + (size_t)row * ldk + k0 + hi * 8;
  v16h a;
#pragma unroll
  for (int g = 0; g < 8; ++g) {
    const int K = (g < 4) ? (2 * g) : (8 + 2 * g);
    a[2 * g]     = r[K];
    a[2 * g + 1] = r[K + 1];
  }
  return a;
}

// B operands pre-swizzled to frag-ready layout: F[((ks*4+t)*32+lane)*16 + i]
static __device__ __forceinline__ v16h wmma_ld_bfrag(const _Float16* F, int ks, int t,
                                                     int lane) {
  const _Float16* fp = F + ((((size_t)ks * 4 + t) * 32 + lane) << 4);
  return *(const v16h*)fp;  // 32B aligned
}

static __device__ __forceinline__ v8f wmma32(v16h a, v16h b, v8f c) {
  return __builtin_amdgcn_wmma_f32_16x16x32_f16(false, a, false, b, (short)0, c,
                                                false, false);
}

// ---------------- float atomic max/min via sign-split int atomics -------------
static __device__ __forceinline__ void atomicMaxF(float* a, float v) {
  if (v >= 0.f) atomicMax((int*)a, __float_as_int(v));
  else          atomicMin((unsigned int*)a, __float_as_uint(v));
}
static __device__ __forceinline__ void atomicMinF(float* a, float v) {
  if (v >= 0.f) atomicMin((int*)a, __float_as_int(v));
  else          atomicMax((unsigned int*)a, __float_as_uint(v));
}

// ---------------- utility kernels ---------------------------------------------
__global__ void k_fill(float* p, float v, size_t n) {
  size_t i  = (size_t)blockIdx.x * blockDim.x + threadIdx.x;
  size_t st = (size_t)gridDim.x * blockDim.x;
  for (; i < n; i += st) p[i] = v;
}

// Swizzle fp32 weight (kmax x 64, row-major) into frag-ready f16 layout.
__global__ void k_prep_b(const float* __restrict__ W, _Float16* __restrict__ F,
                         int kmax, size_t n) {
  size_t tid = (size_t)blockIdx.x * blockDim.x + threadIdx.x;
  if (tid >= n) return;
  const int i    = (int)(tid & 15);
  const int lane = (int)((tid >> 4) & 31);
  const int t    = (int)((tid >> 9) & 3);
  const int ks   = (int)(tid >> 11);
  const int K    = ks * 32 + (lane >> 4) * 8 + ((i < 8) ? i : (8 + i));
  const int nn   = t * 16 + (lane & 15);
  F[tid] = (K < kmax) ? (_Float16)W[(size_t)K * 64 + nn] : (_Float16)0.f;
}

__global__ void k_degree(const int* __restrict__ dstI, float* __restrict__ deg, int E) {
  int i = blockIdx.x * blockDim.x + threadIdx.x;
  if (i < E) atomicAdd(&deg[dstI[i]], 1.0f);
}

// ---------------- fused pretrans GEMM + PNA scatter ---------------------------
static __device__ __forceinline__ void scatter_agg(const v8f* acc, const float* bias,
                                                   const int* dI, float* sum, float* sq,
                                                   float* mx, float* mn, int lane) {
  const int m16 = lane & 15, hi = lane >> 4;
#pragma unroll
  for (int t = 0; t < 4; ++t) {
    const int col = t * 16 + m16;
    const float bb = bias[col];
#pragma unroll
    for (int v = 0; v < 8; ++v) {
      const int d = dI[v + 8 * hi];
      if (d < 0) continue;
      const float val = acc[t][v] + bb;
      const size_t ix = (size_t)d * 64 + col;
      atomicAdd(&sum[ix], val);
      atomicAdd(&sq[ix], val * val);
      atomicMaxF(&mx[ix], val);
      atomicMinF(&mn[ix], val);
    }
  }
}

__global__ __launch_bounds__(256) void k_pretrans(
    const float* __restrict__ h, const float* __restrict__ p, const float* __restrict__ e,
    const int* __restrict__ srcI, const int* __restrict__ dstI,
    const _Float16* __restrict__ FpreH, const float* __restrict__ bPreH,
    const _Float16* __restrict__ FpreP, const float* __restrict__ bPreP,
    float* sum_h, float* sq_h, float* mx_h, float* mn_h,
    float* sum_p, float* sq_p, float* mx_p, float* mn_p) {
  __shared__ _Float16 As[8][16][288];  // per-wave 16-edge A tiles (K padded)
  __shared__ int dIdx[8][16];
  const int w = threadIdx.x >> 5;
  const int lane = threadIdx.x & 31;
  const int ebase = blockIdx.x * 128 + w * 16;
  _Float16* slab = &As[w][0][0];

  // ---- stage h-path rows: [h[src] | p[src] | h[dst] | p[dst] | e | 0-pad] ----
  for (int idx = lane; idx < 16 * 288; idx += 32) {
    const int m = idx / 288, c = idx % 288;
    const int ei = ebase + m;
    float v = 0.f;
    if (ei < EE) {
      const int s = srcI[ei], d = dstI[ei];
      if      (c < 64)  v = h[(size_t)s * 64 + c];
      else if (c < 128) v = p[(size_t)s * 64 + (c - 64)];
      else if (c < 192) v = h[(size_t)d * 64 + (c - 128)];
      else if (c < 256) v = p[(size_t)d * 64 + (c - 192)];
      else if (c < 272) v = e[(size_t)ei * 16 + (c - 256)];
    }
    slab[m * 288 + c] = (_Float16)v;
  }
  if (lane < 16) dIdx[w][lane] = (ebase + lane < EE) ? dstI[ebase + lane] : -1;
  __syncthreads();

  {  // ---- h-path GEMM (K=272 pad 288) + scatter ----
    v8f acc[4] = {};
#pragma unroll
    for (int k0 = 0; k0 < 288; k0 += 32) {
      const int ks = k0 >> 5;
      v16h a = wmma_ld_a_lds(slab, 288, k0, lane);
      acc[0] = wmma32(a, wmma_ld_bfrag(FpreH, ks, 0, lane), acc[0]);
      acc[1] = wmma32(a, wmma_ld_bfrag(FpreH, ks, 1, lane), acc[1]);
      acc[2] = wmma32(a, wmma_ld_bfrag(FpreH, ks, 2, lane), acc[2]);
      acc[3] = wmma32(a, wmma_ld_bfrag(FpreH, ks, 3, lane), acc[3]);
    }
    scatter_agg(acc, bPreH, dIdx[w], sum_h, sq_h, mx_h, mn_h, lane);
  }
  __syncthreads();

  // ---- stage p-path rows: [p[src] | p[dst] | e | 0-pad] (stride 160) ----
  for (int idx = lane; idx < 16 * 160; idx += 32) {
    const int m = idx / 160, c = idx % 160;
    const int ei = ebase + m;
    float v = 0.f;
    if (ei < EE) {
      const int s = srcI[ei], d = dstI[ei];
      if      (c < 64)  v = p[(size_t)s * 64 + c];
      else if (c < 128) v = p[(size_t)d * 64 + (c - 64)];
      else if (c < 144) v = e[(size_t)ei * 16 + (c - 128)];
    }
    slab[m * 160 + c] = (_Float16)v;
  }
  __syncthreads();

  {  // ---- p-path GEMM (K=144 pad 160) + scatter ----
    v8f acc[4] = {};
#pragma unroll
    for (int k0 = 0; k0 < 160; k0 += 32) {
      const int ks = k0 >> 5;
      v16h a = wmma_ld_a_lds(slab, 160, k0, lane);
      acc[0] = wmma32(a, wmma_ld_bfrag(FpreP, ks, 0, lane), acc[0]);
      acc[1] = wmma32(a, wmma_ld_bfrag(FpreP, ks, 1, lane), acc[1]);
      acc[2] = wmma32(a, wmma_ld_bfrag(FpreP, ks, 2, lane), acc[2]);
      acc[3] = wmma32(a, wmma_ld_bfrag(FpreP, ks, 3, lane), acc[3]);
    }
    scatter_agg(acc, bPreP, dIdx[w], sum_p, sq_p, mx_p, mn_p, lane);
  }
}

// ---------------- finalize PNA aggregators -> f16 posttrans A matrices --------
__global__ void k_finalize(const float* __restrict__ deg,
                           const float* sum_h, const float* sq_h,
                           const float* mx_h, const float* mn_h,
                           const float* sum_p, const float* sq_p,
                           const float* mx_p, const float* mn_p,
                           const float* __restrict__ h, const float* __restrict__ p,
                           _Float16* __restrict__ Ah, _Float16* __restrict__ Ap) {
  const size_t tid = (size_t)blockIdx.x * blockDim.x + threadIdx.x;
  if (tid >= (size_t)NN * 64) return;
  const int n = (int)(tid >> 6), c = (int)(tid & 63);
  const float dg = deg[n];
  const float sd = fmaxf(dg, 1.f);
  const bool has = dg > 0.f;
  const float logD = logf(dg + 1.f);
  const float amp = logD / 3.5f;
  const float att = 3.5f / (logD > 0.f ? logD : 1.f);
  const size_t i = tid;
  {
    float mean = sum_h[i] / sd;
    float var = sq_h[i] / sd - mean * mean;  var = var > 0.f ? var : 0.f;
    float stdv = has ? sqrtf(var + 1e-5f) : 0.f;
    float mxv = has ? mx_h[i] : 0.f;
    float mnv = has ? mn_h[i] : 0.f;
    _Float16* r = Ah + (size_t)n * 896;
    r[c]       = (_Float16)h[i];
    r[64 + c]  = (_Float16)p[i];
    r[128 + c] = (_Float16)mean;        r[192 + c] = (_Float16)mxv;
    r[256 + c] = (_Float16)mnv;         r[320 + c] = (_Float16)stdv;
    r[384 + c] = (_Float16)(mean*amp);  r[448 + c] = (_Float16)(mxv*amp);
    r[512 + c] = (_Float16)(mnv*amp);   r[576 + c] = (_Float16)(stdv*amp);
    r[640 + c] = (_Float16)(mean*att);  r[704 + c] = (_Float16)(mxv*att);
    r[768 + c] = (_Float16)(mnv*att);   r[832 + c] = (_Float16)(stdv*att);
  }
  {
    float mean = sum_p[i] / sd;
    float var = sq_p[i] / sd - mean * mean;  var = var > 0.f ? var : 0.f;
    float stdv = has ? sqrtf(var + 1e-5f) : 0.f;
    float mxv = has ? mx_p[i] : 0.f;
    float mnv = has ? mn_p[i] : 0.f;
    _Float16* r = Ap + (size_t)n * 832;
    r[c]       = (_Float16)p[i];
    r[64 + c]  = (_Float16)mean;        r[128 + c] = (_Float16)mxv;
    r[192 + c] = (_Float16)mnv;         r[256 + c] = (_Float16)stdv;
    r[320 + c] = (_Float16)(mean*amp);  r[384 + c] = (_Float16)(mxv*amp);
    r[448 + c] = (_Float16)(mnv*amp);   r[512 + c] = (_Float16)(stdv*amp);
    r[576 + c] = (_Float16)(mean*att);  r[640 + c] = (_Float16)(mxv*att);
    r[704 + c] = (_Float16)(mnv*att);   r[768 + c] = (_Float16)(stdv*att);
  }
}

// ---------------- posttrans GEMM (optionally + snorm + BN stats) --------------
template <bool BN>
__global__ __launch_bounds__(256) void k_post(const _Float16* __restrict__ A, int ldk,
                                              const _Float16* __restrict__ F,
                                              const float* __restrict__ bias,
                                              const float* __restrict__ snorm,
                                              float* __restrict__ out,
                                              float* bnSum, float* bnSq, int nrows) {
  __shared__ float s_sum[64], s_sq[64];
  if (BN) {
    if (threadIdx.x < 64) { s_sum[threadIdx.x] = 0.f; s_sq[threadIdx.x] = 0.f; }
    __syncthreads();
  }
  const int w = threadIdx.x >> 5, lane = threadIdx.x & 31;
  const int m16 = lane & 15, hi = lane >> 4;
  const int row0 = blockIdx.x * 128 + w * 16;
  v8f acc[4] = {};
  for (int k0 = 0; k0 < ldk; k0 += 32) {
    const int ks = k0 >> 5;
    v16h a = wmma_ld_a_g(A, row0, ldk, k0, nrows, lane);
#pragma unroll
    for (int t = 0; t < 4; ++t)
      acc[t] = wmma32(a, wmma_ld_bfrag(F, ks, t, lane), acc[t]);
  }
#pragma unroll
  for (int t = 0; t < 4; ++t) {
    const int col = t * 16 + m16;
    const float bb = bias[col];
#pragma unroll
    for (int v = 0; v < 8; ++v) {
      const int row = row0 + v + 8 * hi;
      if (row < nrows) {
        float val = acc[t][v] + bb;
        if (BN) {
          val *= snorm[row];
          atomicAdd(&s_sum[col], val);
          atomicAdd(&s_sq[col], val * val);
        }
        out[(size_t)row * 64 + col] = val;
      }
    }
  }
  if (BN) {
    __syncthreads();
    if (threadIdx.x < 64) {
      atomicAdd(&bnSum[threadIdx.x], s_sum[threadIdx.x]);
      atomicAdd(&bnSq[threadIdx.x], s_sq[threadIdx.x]);
    }
  }
}

__global__ void k_bnfin(const float* bnSum, const float* bnSq, const float* gamma,
                        const float* beta, float* bnA, float* bnB, int n) {
  const int c = threadIdx.x;
  if (c >= 64) return;
  const float mu = bnSum[c] / (float)n;
  float var = bnSq[c] / (float)n - mu * mu;
  if (var < 0.f) var = 0.f;
  const float inv = gamma[c] * rsqrtf(var + 1e-5f);
  bnA[c] = inv;
  bnB[c] = beta[c] - mu * inv;
}

// ---------------- mixing GEMMs (K=64) + activation + residual -----------------
__global__ __launch_bounds__(256) void k_mix_h(const float* __restrict__ X,
                                               const float* __restrict__ bnA,
                                               const float* __restrict__ bnB,
                                               const _Float16* __restrict__ F,
                                               const float* __restrict__ bias,
                                               const float* __restrict__ hin,
                                               float* __restrict__ out, int nrows) {
  const int w = threadIdx.x >> 5, lane = threadIdx.x & 31;
  const int m16 = lane & 15, hi = lane >> 4;
  const int row0 = blockIdx.x * 128 + w * 16;
  int row = row0 + m16;
  if (row >= nrows) row = 0;
  v8f acc[4] = {};
  for (int k0 = 0; k0 < 64; k0 += 32) {
    const float* r  = X + (size_t)row * 64 + k0 + hi * 8;
    const float* ga = bnA + k0 + hi * 8;
    const float* gb = bnB + k0 + hi * 8;
    v16h a;
#pragma unroll
    for (int g = 0; g < 8; ++g) {
      const int K = (g < 4) ? (2 * g) : (8 + 2 * g);
      a[2*g]   = (_Float16)(r[K]   * ga[K]   + gb[K]);
      a[2*g+1] = (_Float16)(r[K+1] * ga[K+1] + gb[K+1]);
    }
#pragma unroll
    for (int t = 0; t < 4; ++t)
      acc[t] = wmma32(a, wmma_ld_bfrag(F, k0 >> 5, t, lane), acc[t]);
  }
#pragma unroll
  for (int t = 0; t < 4; ++t) {
    const int col = t * 16 + m16;
    const float bb = bias[col];
#pragma unroll
    for (int v = 0; v < 8; ++v) {
      const int rr = row0 + v + 8 * hi;
      if (rr < nrows) {
        float val = acc[t][v] + bb;
        val = val > 0.f ? val : 0.01f * val;  // leaky relu
        out[(size_t)rr * 64 + col] = hin[(size_t)rr * 64 + col] + val;
      }
    }
  }
}

__global__ __launch_bounds__(256) void k_mix_p(const float* __restrict__ X,
                                               const _Float16* __restrict__ F,
                                               const float* __restrict__ bias,
                                               const float* __restrict__ pin,
                                               float* __restrict__ out, int nrows) {
  const int w = threadIdx.x >> 5, lane = threadIdx.x & 31;
  const int m16 = lane & 15, hi = lane >> 4;
  const int row0 = blockIdx.x * 128 + w * 16;
  int row = row0 + m16;
  if (row >= nrows) row = 0;
  v8f acc[4] = {};
  for (int k0 = 0; k0 < 64; k0 += 32) {
    const float* r = X + (size_t)row * 64 + k0 + hi * 8;
    v16h a;
#pragma unroll
    for (int g = 0; g < 8; ++g) {
      const int K = (g < 4) ? (2 * g) : (8 + 2 * g);
      a[2*g]   = (_Float16)r[K];
      a[2*g+1] = (_Float16)r[K+1];
    }
#pragma unroll
    for (int t = 0; t < 4; ++t)
      acc[t] = wmma32(a, wmma_ld_bfrag(F, k0 >> 5, t, lane), acc[t]);
  }
#pragma unroll
  for (int t = 0; t < 4; ++t) {
    const int col = t * 16 + m16;
    const float bb = bias[col];
#pragma unroll
    for (int v = 0; v < 8; ++v) {
      const int rr = row0 + v + 8 * hi;
      if (rr < nrows) {
        const float val = tanhf(acc[t][v] + bb);
        out[(size_t)rr * 64 + col] = pin[(size_t)rr * 64 + col] + val;
      }
    }
  }
}

// ---------------- host orchestration ------------------------------------------
extern "C" void kernel_launch(void* const* d_in, const int* in_sizes, int n_in,
                              void* d_out, int out_size, void* d_ws, size_t ws_size,
                              hipStream_t stream) {
  (void)in_sizes; (void)n_in; (void)out_size; (void)ws_size;
  const float* h      = (const float*)d_in[0];
  const float* p      = (const float*)d_in[1];
  const float* e      = (const float*)d_in[2];
  const float* snorm  = (const float*)d_in[3];
  const int*   srcI   = (const int*)d_in[4];
  const int*   dstI   = (const int*)d_in[5];
  const float* WpreH  = (const float*)d_in[6];
  const float* bPreH  = (const float*)d_in[7];
  const float* WpreP  = (const float*)d_in[8];
  const float* bPreP  = (const float*)d_in[9];
  const float* WpostH = (const float*)d_in[10];
  const float* bPostH = (const float*)d_in[11];
  const float* WpostP = (const float*)d_in[12];
  const float* bPostP = (const float*)d_in[13];
  const float* gamma  = (const float*)d_in[14];
  const float* beta   = (const float*)d_in[15];
  const float* WmixH  = (const float*)d_in[16];
  const float* bMixH  = (const float*)d_in[17];
  const float* WmixP  = (const float*)d_in[18];
  const float* bMixP  = (const float*)d_in[19];
  float* outF = (float*)d_out;

  char* ws = (char*)d_ws;
  size_t off = 0;
  auto alignUp = [](size_t x) { return (x + 255) & ~(size_t)255; };
  auto allocF = [&](size_t n) -> float* {
    float* r = (float*)(ws + off); off = alignUp(off + n * sizeof(float)); return r;
  };
  auto allocH = [&](size_t n) -> _Float16* {
    _Float16* r = (_Float16*)(ws + off); off = alignUp(off + n * sizeof(_Float16)); return r;
  };

  const size_t ND = (size_t)NN * DD;
  float* deg   = allocF(NN);
  float* sum_h = allocF(ND); float* sq_h = allocF(ND);
  float* mx_h  = allocF(ND); float* mn_h = allocF(ND);
  float* sum_p = allocF(ND); float* sq_p = allocF(ND);
  float* mx_p  = allocF(ND); float* mn_p = allocF(ND);
  float* h_ts  = allocF(ND);
  float* p_t   = allocF(ND);
  float* bnSum = allocF(64); float* bnSq = allocF(64);
  float* bnA   = allocF(64); float* bnB  = allocF(64);
  _Float16* FpreH  = allocH(9 * 2048);
  _Float16* FpreP  = allocH(5 * 2048);
  _Float16* FpostH = allocH(28 * 2048);
  _Float16* FpostP = allocH(26 * 2048);
  _Float16* FmixH  = allocH(2 * 2048);
  _Float16* FmixP  = allocH(2 * 2048);
  _Float16* Ah     = allocH((size_t)NN * 896);
  _Float16* Ap     = allocH((size_t)NN * 832);

  // init accumulators (re-done every call: deterministic)
  k_fill<<<512, 256, 0, stream>>>(deg, 0.f, (size_t)NN);
  k_fill<<<2048, 256, 0, stream>>>(sum_h, 0.f, ND);
  k_fill<<<2048, 256, 0, stream>>>(sq_h, 0.f, ND);
  k_fill<<<2048, 256, 0, stream>>>(sum_p, 0.f, ND);
  k_fill<<<2048, 256, 0, stream>>>(sq_p, 0.f, ND);
  k_fill<<<2048, 256, 0, stream>>>(mx_h, -INFINITY, ND);
  k_fill<<<2048, 256, 0, stream>>>(mx_p, -INFINITY, ND);
  k_fill<<<2048, 256, 0, stream>>>(mn_h, INFINITY, ND);
  k_fill<<<2048, 256, 0, stream>>>(mn_p, INFINITY, ND);
  k_fill<<<1, 128, 0, stream>>>(bnSum, 0.f, 128);  // bnSum+bnSq contiguous

  // weight swizzle -> frag-ready f16
  auto prep = [&](const float* W, _Float16* F, int kmax, int nk) {
    const size_t n = (size_t)nk * 2048;
    k_prep_b<<<(int)((n + 255) / 256), 256, 0, stream>>>(W, F, kmax, n);
  };
  prep(WpreH, FpreH, 272, 9);
  prep(WpreP, FpreP, 144, 5);
  prep(WpostH, FpostH, 896, 28);
  prep(WpostP, FpostP, 832, 26);
  prep(WmixH, FmixH, 64, 2);
  prep(WmixP, FmixP, 64, 2);

  k_degree<<<(EE + 255) / 256, 256, 0, stream>>>(dstI, deg, EE);

  k_pretrans<<<EE / 128, 256, 0, stream>>>(h, p, e, srcI, dstI, FpreH, bPreH, FpreP,
                                           bPreP, sum_h, sq_h, mx_h, mn_h, sum_p, sq_p,
                                           mx_p, mn_p);

  k_finalize<<<(int)((ND + 255) / 256), 256, 0, stream>>>(
      deg, sum_h, sq_h, mx_h, mn_h, sum_p, sq_p, mx_p, mn_p, h, p, Ah, Ap);

  const int rowBlocks = (NN + 127) / 128;
  k_post<true><<<rowBlocks, 256, 0, stream>>>(Ah, 896, FpostH, bPostH, snorm, h_ts,
                                              bnSum, bnSq, NN);
  k_post<false><<<rowBlocks, 256, 0, stream>>>(Ap, 832, FpostP, bPostP, nullptr, p_t,
                                               nullptr, nullptr, NN);
  k_bnfin<<<1, 64, 0, stream>>>(bnSum, bnSq, gamma, beta, bnA, bnB, NN);

  k_mix_h<<<rowBlocks, 256, 0, stream>>>(h_ts, bnA, bnB, FmixH, bMixH, h, outF, NN);
  k_mix_p<<<rowBlocks, 256, 0, stream>>>(p_t, FmixP, bMixP, p, outF + ND, NN);
}
